// GraphBasedLSTMClassifier_79594333929611
// MI455X (gfx1250) — matrix operand order repaired and checked by reading, not compiled, not run on hardware
//
#include <hip/hip_runtime.h>
#include <hip/hip_bf16.h>

// Problem constants (from reference)
#define NN 64      // nodes
#define TT 256     // frames
#define FF 13      // input features
#define EE 4032    // edges (complete digraph, no self loops)

typedef __attribute__((ext_vector_type(16))) _Float16 v16h;
typedef __attribute__((ext_vector_type(8)))  float    v8f;

// ---------------- WMMA fragment helpers (gfx1250 wave32 layouts) ----------------
// A fragment: 16x32 f16 tile from row-major [M][K] LDS array (contiguous K -> ds_load_b128).
__device__ __forceinline__ v16h frag_a16(const _Float16* __restrict__ base, int ld,
                                         int r0, int c0, int lane) {
  const int m   = lane & 15;
  const int klo = (lane & 16) ? 8 : 0;
  const _Float16* p = base + (r0 + m) * ld + c0;
  v16h a;
#pragma unroll
  for (int h = 0; h < 8; ++h) a[h] = p[klo + h];
#pragma unroll
  for (int h = 0; h < 8; ++h) a[8 + h] = p[klo + 16 + h];
  return a;
}

// B fragment from a TRANSPOSED store: array is [N][K] row-major (contiguous K -> ds_load_b128).
__device__ __forceinline__ v16h frag_bT16(const _Float16* __restrict__ base, int ld,
                                          int k0, int n0, int lane) {
  const int n  = lane & 15;
  const int kg = (lane & 16) ? 8 : 0;
  const _Float16* p = base + (n0 + n) * ld + k0 + kg;
  v16h b;
#pragma unroll
  for (int h = 0; h < 8; ++h) b[h] = p[h];
#pragma unroll
  for (int h = 0; h < 8; ++h) b[8 + h] = p[16 + h];
  return b;
}

__device__ __forceinline__ v8f wmma_f16(v16h a, v16h b, v8f c) {
  return __builtin_amdgcn_wmma_f32_16x16x32_f16(false, a, false, b, (short)0, c, false, false);
}

// D store, natural [M][N] layout (strided b16 stores)
__device__ __forceinline__ void store_d_bias_relu_f16(_Float16* dst, int ld, int r0, int c0,
                                                      v8f d, const float* __restrict__ bias,
                                                      int lane) {
  const int n  = lane & 15;
  const int mb = (lane & 16) ? 8 : 0;
  const float bv = bias[c0 + n];
#pragma unroll
  for (int v = 0; v < 8; ++v) {
    float val = d[v] + bv;
    dst[(r0 + mb + v) * ld + c0 + n] = (_Float16)(val > 0.f ? val : 0.f);
  }
}

// D store TRANSPOSED: logical D tile (rows r0.., cols c0..) written into [N][M] array
// (8 contiguous halves per lane -> vectorizable LDS store)
__device__ __forceinline__ void store_dT_f16(_Float16* dst, int ldT, int r0, int c0,
                                             v8f d, int lane) {
  const int n  = lane & 15;
  const int mb = (lane & 16) ? 8 : 0;
  _Float16* p = dst + (c0 + n) * ldT + r0 + mb;
#pragma unroll
  for (int v = 0; v < 8; ++v) p[v] = (_Float16)d[v];
}

__device__ __forceinline__ float lrelu02(float x) { return x > 0.f ? x : 0.2f * x; }
__device__ __forceinline__ float sigf(float x) { return 1.f / (1.f + expf(-x)); }

// =====================================================================
// Kernel A: one block per frame. GAT1 -> GAT2 -> attn+max pool -> embs[t]
// =====================================================================
__global__ __launch_bounds__(256) void gnn_frame_kernel(
    const float* __restrict__ x,        // [T][64][13]
    const float* __restrict__ ea_all,   // [T][E]
    const float* __restrict__ W1,       // [13][128]
    const float* __restrict__ a_src1,   // [4][32]
    const float* __restrict__ a_dst1,   // [4][32]
    const float* __restrict__ We1,      // [128]
    const float* __restrict__ a_edge1,  // [4][32]
    const float* __restrict__ b1,       // [128]
    const float* __restrict__ W2,       // [128][128]
    const float* __restrict__ a_src2,   // [128]
    const float* __restrict__ a_dst2,   // [128]
    const float* __restrict__ We2,      // [128]
    const float* __restrict__ a_edge2,  // [128]
    const float* __restrict__ b2,       // [128]
    const float* __restrict__ Wg,       // [128]
    const float* __restrict__ bg,       // [1]
    float* __restrict__ embs)           // [T][256]
{
  __shared__ __align__(16) unsigned char smem[63488];
  // R0 (16KB): H^T [128 ch][64 node]  ->  W2^T chunk [64 out][128 in]  ->  h2 [64][128]
  _Float16* RT   = (_Float16*)(smem);
  // R1 (16KB): {X16 [64][32], W1^T [128][32]} -> h1 [64][128] -> H2pre^T [128 out][64 node]
  _Float16* RD   = (_Float16*)(smem + 16384);
  float*    EA   = (float*)(smem + 32768);      // 4032 f32 edge attrs (async-filled)
  _Float16* COEF = (_Float16*)(smem + 49152);   // 64x64 f16 attention coef
  float*    SS1  = (float*)(smem + 57344);      // [64*4]
  float*    SD1  = (float*)(smem + 58368);      // [64*4]
  float*    AMAX = (float*)(smem + 59392);      // [64*4]
  float*    DEN  = (float*)(smem + 60416);      // [64*4]
  float*    WED  = (float*)(smem + 61440);      // [8]
  float*    SS2  = (float*)(smem + 61472);      // [64]
  float*    SD2  = (float*)(smem + 61728);      // [64]
  float*    GT   = (float*)(smem + 61984);      // [64]
  float*    ATT  = (float*)(smem + 62240);      // [64]

  const int t    = blockIdx.x;
  const int tid  = threadIdx.x;
  const int lane = tid & 31;
  const int wave = tid >> 5;

  const float* xt  = x + t * (NN * FF);
  const float* eat = ea_all + t * EE;
  __builtin_prefetch(W2, 0, 1);  // global_prefetch_b8: warm L2 for the big weight

  // ---- async copy edge attrs straight into LDS (pure f32 copy, 1008 x 16B chunks)
  for (int i = tid; i < EE / 4; i += 256) {
    unsigned loff = (unsigned)(unsigned long long)(uintptr_t)(smem + 32768 + i * 16);
    const float* gp = eat + i * 4;
    asm volatile("global_load_async_to_lds_b128 %0, %1, off"
                 :: "v"(loff), "v"(gp) : "memory");
  }

  // ---- stage x (pad K 13->32) and W1 transposed [128 p][32 k] as f16, overlapped with async
  _Float16* X16 = RD;            // 64x32
  _Float16* W1T = RD + 64 * 32;  // 128x32
  for (int i = tid; i < 64 * 32; i += 256) {
    int n = i >> 5, k = i & 31;
    X16[i] = (_Float16)((k < FF) ? xt[n * FF + k] : 0.f);
  }
  for (int i = tid; i < 128 * 32; i += 256) {
    int p = i >> 5, k = i & 31;
    W1T[i] = (_Float16)((k < FF) ? W1[k * 128 + p] : 0.f);
  }
  asm volatile("s_wait_asynccnt 0x0" ::: "memory");
  __syncthreads();

  // ---- GEMM1: H(64x128) = X(64x32) @ W1(32x128); D stored transposed into RT
#pragma unroll
  for (int q = 0; q < 4; ++q) {
    int tix = wave * 4 + q, mt = tix >> 3, nt = tix & 7;
    v16h a = frag_a16(X16, 32, mt * 16, 0, lane);
    v16h b = frag_bT16(W1T, 32, 0, nt * 16, lane);
    v8f  c = {};
    c = wmma_f16(a, b, c);
    store_dT_f16(RT, 64, mt * 16, nt * 16, c, lane);   // H^T[ch][node]
  }
  __syncthreads();

  // ---- per-node attention dots, layer 1 (+ edge constants)
  {
    int n = tid >> 2, hd = tid & 3;
    float ss = 0.f, sd = 0.f;
#pragma unroll
    for (int c = 0; c < 32; ++c) {
      float hv = (float)RT[(hd * 32 + c) * 64 + n];
      ss += hv * a_src1[hd * 32 + c];
      sd += hv * a_dst1[hd * 32 + c];
    }
    SS1[n * 4 + hd] = ss;
    SD1[n * 4 + hd] = sd;
    if (tid < 4) {
      float w = 0.f;
      for (int c = 0; c < 32; ++c) w += We1[tid * 32 + c] * a_edge1[tid * 32 + c];
      WED[tid] = w;
    }
    if (tid == 4) {
      float w = 0.f;
      for (int c = 0; c < 128; ++c) w += We2[c] * a_edge2[c];
      WED[4] = w;
    }
  }
  __syncthreads();

  // ---- per-dst softmax stats, layer 1: thread = (j,hd)
  {
    int j = tid >> 2, hd = tid & 3;
    float sdj = SD1[j * 4 + hd], wd = WED[hd];
    float m = -1e30f;
    for (int i = 0; i < 64; ++i) {
      if (i == j) continue;
      int e = i * 63 + (j > i ? j - 1 : j);
      m = fmaxf(m, lrelu02(SS1[i * 4 + hd] + sdj + EA[e] * wd));
    }
    float s = 0.f;
    for (int i = 0; i < 64; ++i) {
      if (i == j) continue;
      int e = i * 63 + (j > i ? j - 1 : j);
      s += expf(lrelu02(SS1[i * 4 + hd] + sdj + EA[e] * wd) - m);
    }
    AMAX[tid] = m; DEN[tid] = s;
  }
  __syncthreads();

  // ---- aggregation, layer 1: per head, OUT_head(64x32) = coef(64x64) @ H_head
  v8f acc1[4];
#pragma unroll
  for (int hd = 0; hd < 4; ++hd) { v8f z = {}; acc1[hd] = z; }
#pragma unroll
  for (int hd = 0; hd < 4; ++hd) {
    { // build coef matrix for this head
      int j = tid >> 2, iq = tid & 3;
      float am = AMAX[j * 4 + hd], dn = DEN[j * 4 + hd] + 1e-16f;
      float sdj = SD1[j * 4 + hd], wd = WED[hd];
#pragma unroll
      for (int ii = 0; ii < 16; ++ii) {
        int i = iq * 16 + ii;
        float cfv = 0.f;
        if (i != j) {
          int e = i * 63 + (j > i ? j - 1 : j);
          cfv = expf(lrelu02(SS1[i * 4 + hd] + sdj + EA[e] * wd) - am) / dn;
        }
        COEF[j * 64 + i] = (_Float16)cfv;
      }
    }
    __syncthreads();
    {
      int mt = wave >> 1, nt = wave & 1;
#pragma unroll
      for (int kt = 0; kt < 2; ++kt) {
        v16h a = frag_a16(COEF, 64, mt * 16, kt * 32, lane);
        v16h b = frag_bT16(RT, 64, kt * 32, hd * 32 + nt * 16, lane); // B = H (K=node) from H^T
        acc1[hd] = wmma_f16(a, b, acc1[hd]);
      }
    }
    __syncthreads();
  }
  { // h1 = relu(agg + b1) -> RD, natural [node][ch] (serves as GEMM2 A operand)
    int mt = wave >> 1, nt = wave & 1;
#pragma unroll
    for (int hd = 0; hd < 4; ++hd)
      store_d_bias_relu_f16(RD, 128, mt * 16, hd * 32 + nt * 16, acc1[hd], b1, lane);
  }
  __syncthreads();

  // ---- GEMM2: H2pre(64x128) = h1(64x128) @ W2(128x128)
  // B = W2^T staged in two 64-out-channel chunks (full K per chunk -> no cross-chunk accum)
  v8f acc2[4];
#pragma unroll
  for (int q = 0; q < 4; ++q) { v8f z = {}; acc2[q] = z; }
#pragma unroll
  for (int cc = 0; cc < 2; ++cc) {
    for (int i = tid; i < 64 * 128; i += 256) {
      int ol = i >> 7, ic = i & 127;                      // local out-ch, in-ch
      RT[i] = (_Float16)W2[ic * 128 + (cc * 64 + ol)];    // W2^T[out][in]
    }
    __syncthreads();
#pragma unroll
    for (int q = 0; q < 2; ++q) {
      int tix = wave * 2 + q, mt = tix >> 2, ntl = tix & 3;
#pragma unroll
      for (int ks = 0; ks < 4; ++ks) {
        v16h a = frag_a16(RD, 128, mt * 16, ks * 32, lane);
        v16h b = frag_bT16(RT, 128, ks * 32, ntl * 16, lane);
        acc2[cc * 2 + q] = wmma_f16(a, b, acc2[cc * 2 + q]);
      }
    }
    __syncthreads();
  }
#pragma unroll
  for (int cc = 0; cc < 2; ++cc)   // store H2pre^T[out][node] into RD (h1 is dead)
#pragma unroll
    for (int q = 0; q < 2; ++q) {
      int tix = wave * 2 + q, mt = tix >> 2, ntl = tix & 3;
      store_dT_f16(RD, 64, mt * 16, cc * 64 + ntl * 16, acc2[cc * 2 + q], lane);
    }
  __syncthreads();

  // ---- attention dots, layer 2 (read H2pre^T)
  if (tid < 128) {
    int n = tid & 63;
    float s = 0.f;
    if (tid < 64) {
      for (int c = 0; c < 128; ++c) s += (float)RD[c * 64 + n] * a_src2[c];
      SS2[n] = s;
    } else {
      for (int c = 0; c < 128; ++c) s += (float)RD[c * 64 + n] * a_dst2[c];
      SD2[n] = s;
    }
  }
  __syncthreads();
  if (tid < 64) {
    int j = tid;
    float sdj = SD2[j], wd = WED[4];
    float m = -1e30f;
    for (int i = 0; i < 64; ++i) {
      if (i == j) continue;
      int e = i * 63 + (j > i ? j - 1 : j);
      m = fmaxf(m, lrelu02(SS2[i] + sdj + EA[e] * wd));
    }
    float s = 0.f;
    for (int i = 0; i < 64; ++i) {
      if (i == j) continue;
      int e = i * 63 + (j > i ? j - 1 : j);
      s += expf(lrelu02(SS2[i] + sdj + EA[e] * wd) - m);
    }
    AMAX[j] = m; DEN[j] = s;
  }
  __syncthreads();
  { // coef2 (64x64 f16)
    int j = tid >> 2, iq = tid & 3;
    float am = AMAX[j], dn = DEN[j] + 1e-16f;
    float sdj = SD2[j], wd = WED[4];
#pragma unroll
    for (int ii = 0; ii < 16; ++ii) {
      int i = iq * 16 + ii;
      float cfv = 0.f;
      if (i != j) {
        int e = i * 63 + (j > i ? j - 1 : j);
        cfv = expf(lrelu02(SS2[i] + sdj + EA[e] * wd) - am) / dn;
      }
      COEF[j * 64 + i] = (_Float16)cfv;
    }
  }
  __syncthreads();

  // ---- aggregation, layer 2: OUT(64x128) = coef2(64x64) @ H2pre(64x128)
  v8f acc3[4];
#pragma unroll
  for (int q = 0; q < 4; ++q) { v8f z = {}; acc3[q] = z; }
#pragma unroll
  for (int q = 0; q < 4; ++q) {
    int tix = wave * 4 + q, mt = tix >> 3, nt = tix & 7;
#pragma unroll
    for (int kt = 0; kt < 2; ++kt) {
      v16h a = frag_a16(COEF, 64, mt * 16, kt * 32, lane);
      v16h b = frag_bT16(RD, 64, kt * 32, nt * 16, lane); // B = H2pre (K=node) from H2pre^T
      acc3[q] = wmma_f16(a, b, acc3[q]);
    }
  }
  __syncthreads();
#pragma unroll
  for (int q = 0; q < 4; ++q) { // h2 = relu(agg + b2) -> RT [node][ch]
    int tix = wave * 4 + q, mt = tix >> 3, nt = tix & 7;
    store_d_bias_relu_f16(RT, 128, mt * 16, nt * 16, acc3[q], b2, lane);
  }
  __syncthreads();

  // ---- attention pooling + max pooling -> embs[t][0:256]
  if (tid < 64) {
    float g = bg[0];
    for (int c = 0; c < 128; ++c) g += (float)RT[tid * 128 + c] * Wg[c];
    GT[tid] = g;
  }
  __syncthreads();
  if (tid < 64) {
    float m = -1e30f;
    for (int k = 0; k < 64; ++k) m = fmaxf(m, GT[k]);
    float s = 0.f;
    for (int k = 0; k < 64; ++k) s += expf(GT[k] - m);
    ATT[tid] = expf(GT[tid] - m) / s;
  }
  __syncthreads();
  if (tid < 128) {
    int c = tid;
    float ps = 0.f, mx = -1e30f;
    for (int n = 0; n < 64; ++n) {
      float hv = (float)RT[n * 128 + c];
      ps += ATT[n] * hv;
      mx = fmaxf(mx, hv);
    }
    embs[t * 256 + c] = ps;
    embs[t * 256 + 128 + c] = mx;
  }
}

// =====================================================================
// Kernel B: Z = embs(256x256) @ [Wih_f; Wih_b].T + (bih+bhh)  via WMMA
// grid (16,4): 64x64 output tiles over a 256x1024 result
// =====================================================================
__global__ __launch_bounds__(256) void lstm_in_gemm_kernel(
    const float* __restrict__ embs,   // [256][256]
    const float* __restrict__ Wih_f,  // [512][256]
    const float* __restrict__ bih_f, const float* __restrict__ bhh_f,
    const float* __restrict__ Wih_b,  // [512][256]
    const float* __restrict__ bih_b, const float* __restrict__ bhh_b,
    float* __restrict__ Zf, float* __restrict__ Zb)  // each [256][512]
{
  __shared__ _Float16 Ae[64 * 128];  // embs chunk [64 rows][128 k]
  __shared__ _Float16 Bw[64 * 128];  // W rows (N) x k chunk  ([N][K], contiguous K)
  const int tid = threadIdx.x, lane = tid & 31, wave = tid >> 5;
  const int bn = blockIdx.x;   // 0..15 (64-col tile of the 1024 gate cols)
  const int bm = blockIdx.y;   // 0..3  (64-row tile of 256 timesteps)

  v8f acc[2];
  { v8f z = {}; acc[0] = z; acc[1] = z; }
#pragma unroll
  for (int kh = 0; kh < 2; ++kh) {
    for (int i = tid; i < 64 * 128; i += 256) {
      int r = i >> 7, c = i & 127;
      Ae[i] = (_Float16)embs[(bm * 64 + r) * 256 + kh * 128 + c];
      int g = bn * 64 + r;
      const float* W = (g < 512) ? (Wih_f + g * 256) : (Wih_b + (g - 512) * 256);
      Bw[i] = (_Float16)W[kh * 128 + c];
    }
    __syncthreads();
#pragma unroll
    for (int q = 0; q < 2; ++q) {
      int st = wave * 2 + q, mt = st >> 2, nt = st & 3;
#pragma unroll
      for (int ks = 0; ks < 4; ++ks) {
        v16h a = frag_a16(Ae, 128, mt * 16, ks * 32, lane);
        v16h b = frag_bT16(Bw, 128, ks * 32, nt * 16, lane);
        acc[q] = wmma_f16(a, b, acc[q]);
      }
    }
    __syncthreads();
  }
#pragma unroll
  for (int q = 0; q < 2; ++q) {
    int st = wave * 2 + q, mt = st >> 2, nt = st & 3;
    int n = lane & 15, mb = (lane & 16) ? 8 : 0;
    int colg = bn * 64 + nt * 16 + n;
    float bias = (colg < 512) ? (bih_f[colg] + bhh_f[colg])
                              : (bih_b[colg - 512] + bhh_b[colg - 512]);
#pragma unroll
    for (int v = 0; v < 8; ++v) {
      int row = bm * 64 + mt * 16 + mb + v;
      float val = acc[q][v] + bias;
      if (colg < 512) Zf[row * 512 + colg] = val;
      else            Zb[row * 512 + (colg - 512)] = val;
    }
  }
}

// =====================================================================
// Kernel C: forward LSTM over 256 steps + single backward step + output.
// (outs_b[0] only needs one bwd step from zero state on embs[T-1].)
// =====================================================================
__global__ __launch_bounds__(512) void lstm_seq_kernel(
    const float* __restrict__ Zf, const float* __restrict__ Zb,
    const float* __restrict__ Whh_f,   // [512][128]
    const float* __restrict__ Wo,      // [256]
    const float* __restrict__ bo,      // [1]
    float* __restrict__ out)
{
  __shared__ float hf[128], cf[128], gate[512], hb[128];
  const int tid = threadIdx.x;
  if (tid < 128) { hf[tid] = 0.f; cf[tid] = 0.f; }
  __syncthreads();

  for (int t = 0; t < 256; ++t) {
    float g = Zf[t * 512 + tid];
    const float* w = Whh_f + tid * 128;
#pragma unroll 8
    for (int k = 0; k < 128; ++k) g += w[k] * hf[k];
    gate[tid] = g;
    __syncthreads();
    if (tid < 128) {
      float gi = gate[tid], gfo = gate[128 + tid], gg = gate[256 + tid], go = gate[384 + tid];
      float c = sigf(gfo) * cf[tid] + sigf(gi) * tanhf(gg);
      cf[tid] = c;
      hf[tid] = sigf(go) * tanhf(c);
    }
    __syncthreads();
  }

  if (tid < 128) {  // backward LSTM, first step only (h0=c0=0 -> no Whh term)
    float gi = Zb[255 * 512 + tid];
    float gg = Zb[255 * 512 + 256 + tid];
    float go = Zb[255 * 512 + 384 + tid];
    float c  = sigf(gi) * tanhf(gg);
    hb[tid]  = sigf(go) * tanhf(c);
  }
  __syncthreads();

  if (tid == 0) {
    float s = bo[0];
    for (int k = 0; k < 128; ++k) s += hf[k] * Wo[k];
    for (int k = 0; k < 128; ++k) s += hb[k] * Wo[128 + k];
    out[0] = sigf(s);
  }
}

// =====================================================================
extern "C" void kernel_launch(void* const* d_in, const int* in_sizes, int n_in,
                              void* d_out, int out_size, void* d_ws, size_t ws_size,
                              hipStream_t stream) {
  (void)in_sizes; (void)n_in; (void)out_size; (void)ws_size;
  const float* x       = (const float*)d_in[0];
  // d_in[1] = edge_index (unused: complete-graph structure is analytic)
  const float* ea      = (const float*)d_in[2];
  const float* W1      = (const float*)d_in[3];
  const float* a_src1  = (const float*)d_in[4];
  const float* a_dst1  = (const float*)d_in[5];
  const float* We1     = (const float*)d_in[6];
  const float* a_edge1 = (const float*)d_in[7];
  const float* b1      = (const float*)d_in[8];
  const float* W2      = (const float*)d_in[9];
  const float* a_src2  = (const float*)d_in[10];
  const float* a_dst2  = (const float*)d_in[11];
  const float* We2     = (const float*)d_in[12];
  const float* a_edge2 = (const float*)d_in[13];
  const float* b2      = (const float*)d_in[14];
  const float* Wg      = (const float*)d_in[15];
  const float* bg      = (const float*)d_in[16];
  const float* Wih_f   = (const float*)d_in[17];
  const float* Whh_f   = (const float*)d_in[18];
  const float* bih_f   = (const float*)d_in[19];
  const float* bhh_f   = (const float*)d_in[20];
  const float* Wih_b   = (const float*)d_in[21];
  // d_in[22] = Whh_b (unused: only first bwd step needed, h0 = 0)
  const float* bih_b   = (const float*)d_in[23];
  const float* bhh_b   = (const float*)d_in[24];
  const float* Wo      = (const float*)d_in[25];
  const float* bo      = (const float*)d_in[26];

  float* ws   = (float*)d_ws;
  float* embs = ws;                          // 256*256 floats
  float* Zf   = ws + 256 * 256;              // 256*512 floats
  float* Zb   = Zf + 256 * 512;              // 256*512 floats

  gnn_frame_kernel<<<TT, 256, 0, stream>>>(x, ea, W1, a_src1, a_dst1, We1, a_edge1, b1,
                                           W2, a_src2, a_dst2, We2, a_edge2, b2, Wg, bg,
                                           embs);
  lstm_in_gemm_kernel<<<dim3(16, 4), 256, 0, stream>>>(embs, Wih_f, bih_f, bhh_f,
                                                       Wih_b, bih_b, bhh_b, Zf, Zb);
  lstm_seq_kernel<<<1, 512, 0, stream>>>(Zf, Zb, Whh_f, Wo, bo, (float*)d_out);
}
